// GatedGCNLayer_36180804502137
// MI455X (gfx1250) — matrix operand order repaired and checked by reading, not compiled, not run on hardware
//
#include <hip/hip_runtime.h>
#include <hip/hip_bf16.h>

// ---------------------------------------------------------------------------
// GatedGCN layer for MI455X (gfx1250), wave32, WMMA bf16 path.
// ---------------------------------------------------------------------------

#define DFEAT 128
#define NNODE 50000
#define NEDGE 640000
#define ND_ELEMS (NNODE * DFEAT)      // 6,400,000
#define ED_ELEMS (NEDGE * DFEAT)      // 81,920,000
#define BN_EPS 1e-5f

typedef __attribute__((ext_vector_type(16))) __bf16        v16bf;
typedef __attribute__((ext_vector_type(2)))  __bf16        v2bf;
typedef __attribute__((ext_vector_type(8)))  float         v8f;
typedef __attribute__((ext_vector_type(8)))  unsigned int  v8u;

// fp32 pair -> packed bf16 (RNE). Prefer the hardware packed convert.
__device__ __forceinline__ unsigned int pack2bf(float lo, float hi) {
#if __has_builtin(__builtin_amdgcn_cvt_pk_bf16_f32)
  v2bf p = __builtin_amdgcn_cvt_pk_bf16_f32(lo, hi);
  return __builtin_bit_cast(unsigned int, p);
#else
  v2bf p;
  p[0] = (__bf16)lo;          // fptrunc -> v_cvt_*_bf16_f32 on gfx1250
  p[1] = (__bf16)hi;
  return __builtin_bit_cast(unsigned int, p);
#endif
}

// Device-scope hardware f32 atomic add (no return -> STOREcnt only, drained by
// the implicit wait in S_ENDPGM; cross-kernel ordering comes from the stream).
// Forced via inline asm so the scatter can never fall back to a CAS loop.
__device__ __forceinline__ void gatomic_add(float* p, float v) {
  asm volatile("global_atomic_add_f32 %0, %1, off scope:SCOPE_DEV"
               :
               : "v"(p), "v"(v)
               : "memory");
}
__device__ __forceinline__ void latomic_add(float* p, float v) {
  __hip_atomic_fetch_add(p, v, __ATOMIC_RELAXED, __HIP_MEMORY_SCOPE_WORKGROUP);
}

// ---------------------------------------------------------------------------
// Zero fill
// ---------------------------------------------------------------------------
__global__ void k_zero(float* __restrict__ p, int n) {
  int i = blockIdx.x * blockDim.x + threadIdx.x;
  if (i < n) p[i] = 0.0f;
}

// ---------------------------------------------------------------------------
// H = clamp_rows(X) @ W^T + b        (X:[nrows,128] f32, W:[128,128] f32)
// Block = 256 thr (8 waves). Wave owns 16 rows; 8 N-tiles x 4 K-steps WMMA.
// W staged in LDS pre-packed into B-fragment pair layout:
//   sWp[n*65 + k/2] = pack_bf16(W[n][k], W[n][k+1])   (65 pad => no bank conflict)
// ---------------------------------------------------------------------------
__global__ __launch_bounds__(256) void k_gemm_xwt(
    const float* __restrict__ X, const float* __restrict__ W,
    const float* __restrict__ bias, float* __restrict__ H, int nrows)
{
  __shared__ unsigned int sWp[128 * 65];
  const int t = threadIdx.x;
  for (int idx = t; idx < 128 * 64; idx += 256) {
    int n  = idx >> 6;          // n 0..127
    int kp = idx & 63;          // k-pair 0..63
    sWp[n * 65 + kp] = pack2bf(W[n * DFEAT + 2 * kp],
                               W[n * DFEAT + 2 * kp + 1]);
  }
  __syncthreads();

  const int lane = t & 31;
  const int wave = t >> 5;
  const int hi   = lane >> 4;       // half-wave select
  const int ln   = lane & 15;
  const int rBase = blockIdx.x * 128 + wave * 16;

  // A fragments (16-bit A 16x32 layout), all 4 K-steps resident in VGPRs.
  v8u afr[4];
  {
    int row = rBase + ln;
    if (row >= nrows) row = nrows - 1;            // clamp: EXEC stays all-ones
    const float* xr = X + (size_t)row * DFEAT;
#pragma unroll
    for (int s = 0; s < 4; ++s) {
#pragma unroll
      for (int v = 0; v < 8; ++v) {
        int k = 32 * s + 2 * v + 16 * hi;
        afr[s][v] = pack2bf(xr[k], xr[k + 1]);
      }
    }
  }

  const int kk = ln + 16 * hi;     // B-fragment K within a 32-step
#pragma unroll
  for (int nt = 0; nt < 8; ++nt) {
    float bv = bias[nt * 16 + ln];
    v8f acc;
#pragma unroll
    for (int r = 0; r < 8; ++r) acc[r] = bv;
#pragma unroll
    for (int s = 0; s < 4; ++s) {
      v8u bfr;
      const unsigned int* bp = &sWp[(nt * 16 + ln) * 65 + 16 * s + 8 * hi];
#pragma unroll
      for (int v = 0; v < 8; ++v) bfr[v] = bp[v];
      acc = __builtin_amdgcn_wmma_f32_16x16x32_bf16(
          false, __builtin_bit_cast(v16bf, afr[s]),
          false, __builtin_bit_cast(v16bf, bfr),
          (short)0, acc, false, false);
    }
#pragma unroll
    for (int r = 0; r < 8; ++r) {
      int row = rBase + r + 8 * hi;
      if (row < nrows) H[(size_t)row * DFEAT + nt * 16 + ln] = acc[r];
    }
  }
}

// ---------------------------------------------------------------------------
// edge_in = edge_attr @ WA^T + bA + Bx[row] + Cx[col]; accumulate BN stats.
// 128 edges / block (E = 5000 * 128 exactly).
// ---------------------------------------------------------------------------
__global__ __launch_bounds__(256) void k_edge_gemm(
    const float* __restrict__ EA, const float* __restrict__ WA,
    const float* __restrict__ bA, const long long* __restrict__ eidx,
    const float* __restrict__ Bx, const float* __restrict__ Cx,
    float* __restrict__ edge_in, float* __restrict__ gSum,
    float* __restrict__ gSq)
{
  __shared__ unsigned int sWp[128 * 65];
  __shared__ int   sRow[128], sCol[128];
  __shared__ float bSum[128], bSq[128];
  const int t = threadIdx.x;
  for (int idx = t; idx < 128 * 64; idx += 256) {
    int n  = idx >> 6;
    int kp = idx & 63;
    sWp[n * 65 + kp] = pack2bf(WA[n * DFEAT + 2 * kp],
                               WA[n * DFEAT + 2 * kp + 1]);
  }
  if (t < 128) {
    int e = blockIdx.x * 128 + t;
    sRow[t] = (int)eidx[e];
    sCol[t] = (int)eidx[NEDGE + e];
    bSum[t] = 0.0f;
    bSq[t]  = 0.0f;
  }
  __syncthreads();

  const int lane = t & 31;
  const int wave = t >> 5;
  const int hi   = lane >> 4;
  const int ln   = lane & 15;
  const int eBase = blockIdx.x * 128 + wave * 16;

  v8u afr[4];
  {
    const float* er = EA + (size_t)(eBase + ln) * DFEAT;
#pragma unroll
    for (int s = 0; s < 4; ++s) {
#pragma unroll
      for (int v = 0; v < 8; ++v) {
        int k = 32 * s + 2 * v + 16 * hi;
        afr[s][v] = pack2bf(er[k], er[k + 1]);
      }
    }
  }

  const int kk = ln + 16 * hi;
#pragma unroll
  for (int nt = 0; nt < 8; ++nt) {
    const int c = nt * 16 + ln;
    float bv = bA[c];
    v8f acc;
#pragma unroll
    for (int r = 0; r < 8; ++r) acc[r] = bv;
#pragma unroll
    for (int s = 0; s < 4; ++s) {
      v8u bfr;
      const unsigned int* bp = &sWp[(nt * 16 + ln) * 65 + 16 * s + 8 * hi];
#pragma unroll
      for (int v = 0; v < 8; ++v) bfr[v] = bp[v];
      acc = __builtin_amdgcn_wmma_f32_16x16x32_bf16(
          false, __builtin_bit_cast(v16bf, afr[s]),
          false, __builtin_bit_cast(v16bf, bfr),
          (short)0, acc, false, false);
    }
    float sAcc = 0.0f, qAcc = 0.0f;
#pragma unroll
    for (int r = 0; r < 8; ++r) {
      int m = wave * 16 + r + 8 * hi;          // edge slot within block
      int e = blockIdx.x * 128 + m;
      float val = acc[r] + Bx[(size_t)sRow[m] * DFEAT + c]
                         + Cx[(size_t)sCol[m] * DFEAT + c];
      edge_in[(size_t)e * DFEAT + c] = val;
      sAcc += val;
      qAcc += val * val;
    }
    latomic_add(&bSum[c], sAcc);
    latomic_add(&bSq[c],  qAcc);
  }
  __syncthreads();
  if (t < 128) {
    gatomic_add(&gSum[t], bSum[t]);
    gatomic_add(&gSq[t],  bSq[t]);
  }
}

// ---------------------------------------------------------------------------
// Fold BN stats into per-feature scale/shift. 1 block, 128 threads.
// ---------------------------------------------------------------------------
__global__ void k_bn_finalize(const float* __restrict__ gSum,
                              const float* __restrict__ gSq,
                              const float* __restrict__ gamma,
                              const float* __restrict__ beta,
                              float cnt_inv, float* __restrict__ scale,
                              float* __restrict__ shift)
{
  int t = threadIdx.x;
  float mu  = gSum[t] * cnt_inv;
  float var = gSq[t] * cnt_inv - mu * mu;
  float rs  = rsqrtf(var + BN_EPS);
  float sc  = gamma[t] * rs;
  scale[t] = sc;
  shift[t] = beta[t] - mu * sc;
}

// ---------------------------------------------------------------------------
// edge_out = edge_attr + relu(bn(edge_in)); msg = sigmoid(edge_out)*Vx[col];
// agg[row] += msg. 8 edges/block, 1 wave per edge, float4 per lane.
// ---------------------------------------------------------------------------
__global__ __launch_bounds__(256) void k_edge_apply(
    const float* __restrict__ edge_in, const float* __restrict__ EA,
    const long long* __restrict__ eidx, const float* __restrict__ scale,
    const float* __restrict__ shift, const float* __restrict__ Vx,
    float* __restrict__ edge_out, float* __restrict__ agg)
{
  int t = threadIdx.x;
  int e = blockIdx.x * 8 + (t >> 5);
  int c = (t & 31) * 4;
  int row = (int)eidx[e];
  int col = (int)eidx[NEDGE + e];
  float4 h  = *(const float4*)(edge_in + (size_t)e * DFEAT + c);
  float4 ea = *(const float4*)(EA      + (size_t)e * DFEAT + c);
  float4 sc = *(const float4*)(scale + c);
  float4 sh = *(const float4*)(shift + c);
  float4 vx = *(const float4*)(Vx + (size_t)col * DFEAT + c);
  float hv[4]  = {h.x, h.y, h.z, h.w};
  float eav[4] = {ea.x, ea.y, ea.z, ea.w};
  float scv[4] = {sc.x, sc.y, sc.z, sc.w};
  float shv[4] = {sh.x, sh.y, sh.z, sh.w};
  float vxv[4] = {vx.x, vx.y, vx.z, vx.w};
  float eo[4];
#pragma unroll
  for (int i = 0; i < 4; ++i) {
    float tv = fmaxf(scv[i] * hv[i] + shv[i], 0.0f);
    eo[i] = eav[i] + tv;
  }
  *(float4*)(edge_out + (size_t)e * DFEAT + c) =
      make_float4(eo[0], eo[1], eo[2], eo[3]);
  float* aggp = agg + (size_t)row * DFEAT + c;
#pragma unroll
  for (int i = 0; i < 4; ++i) {
    float sig = 1.0f / (1.0f + __expf(-eo[i]));
    gatomic_add(aggp + i, sig * vxv[i]);
  }
}

// ---------------------------------------------------------------------------
// Node BN stats over node_in = Ux + agg. 8 rows/block (N = 6250*8 exactly).
// ---------------------------------------------------------------------------
__global__ __launch_bounds__(256) void k_node_stats(
    const float* __restrict__ Ux, const float* __restrict__ agg,
    float* __restrict__ gSum, float* __restrict__ gSq)
{
  __shared__ float bSum[128], bSq[128];
  int t = threadIdx.x;
  if (t < 128) { bSum[t] = 0.0f; bSq[t] = 0.0f; }
  __syncthreads();
  int row = blockIdx.x * 8 + (t >> 5);
  int c = (t & 31) * 4;
  float4 u = *(const float4*)(Ux  + (size_t)row * DFEAT + c);
  float4 a = *(const float4*)(agg + (size_t)row * DFEAT + c);
  float ni[4] = {u.x + a.x, u.y + a.y, u.z + a.z, u.w + a.w};
#pragma unroll
  for (int i = 0; i < 4; ++i) {
    latomic_add(&bSum[c + i], ni[i]);
    latomic_add(&bSq[c + i],  ni[i] * ni[i]);
  }
  __syncthreads();
  if (t < 128) {
    gatomic_add(&gSum[t], bSum[t]);
    gatomic_add(&gSq[t],  bSq[t]);
  }
}

// ---------------------------------------------------------------------------
// x_out = x + relu(bn(Ux + agg))
// ---------------------------------------------------------------------------
__global__ __launch_bounds__(256) void k_node_apply(
    const float* __restrict__ x, const float* __restrict__ Ux,
    const float* __restrict__ agg, const float* __restrict__ scale,
    const float* __restrict__ shift, float* __restrict__ xout)
{
  int t = threadIdx.x;
  int row = blockIdx.x * 8 + (t >> 5);
  int c = (t & 31) * 4;
  float4 xv = *(const float4*)(x   + (size_t)row * DFEAT + c);
  float4 u  = *(const float4*)(Ux  + (size_t)row * DFEAT + c);
  float4 a  = *(const float4*)(agg + (size_t)row * DFEAT + c);
  float4 sc = *(const float4*)(scale + c);
  float4 sh = *(const float4*)(shift + c);
  float xi[4]  = {xv.x, xv.y, xv.z, xv.w};
  float ni[4]  = {u.x + a.x, u.y + a.y, u.z + a.z, u.w + a.w};
  float scv[4] = {sc.x, sc.y, sc.z, sc.w};
  float shv[4] = {sh.x, sh.y, sh.z, sh.w};
  float o[4];
#pragma unroll
  for (int i = 0; i < 4; ++i)
    o[i] = xi[i] + fmaxf(scv[i] * ni[i] + shv[i], 0.0f);
  *(float4*)(xout + (size_t)row * DFEAT + c) =
      make_float4(o[0], o[1], o[2], o[3]);
}

// ---------------------------------------------------------------------------
// Host-side orchestration
// ---------------------------------------------------------------------------
extern "C" void kernel_launch(void* const* d_in, const int* in_sizes, int n_in,
                              void* d_out, int out_size, void* d_ws,
                              size_t ws_size, hipStream_t stream) {
  const float*     x    = (const float*)d_in[0];
  const long long* eidx = (const long long*)d_in[1];
  const float*     ea   = (const float*)d_in[2];
  const float* Wu = (const float*)d_in[3];
  const float* bu = (const float*)d_in[4];
  const float* Wv = (const float*)d_in[5];
  const float* bv = (const float*)d_in[6];
  const float* WA = (const float*)d_in[7];
  const float* bA = (const float*)d_in[8];
  const float* WB = (const float*)d_in[9];
  const float* bB = (const float*)d_in[10];
  const float* WC = (const float*)d_in[11];
  const float* bC = (const float*)d_in[12];
  const float* gn_gamma = (const float*)d_in[13];
  const float* gn_beta  = (const float*)d_in[14];
  const float* ge_gamma = (const float*)d_in[15];
  const float* ge_beta  = (const float*)d_in[16];

  float* ws   = (float*)d_ws;
  float* Ux   = ws;                 // [N*D]
  float* Vx   = Ux + ND_ELEMS;      // [N*D]
  float* Bx   = Vx + ND_ELEMS;      // [N*D]
  float* Cx   = Bx + ND_ELEMS;      // [N*D]
  float* agg  = Cx + ND_ELEMS;      // [N*D]  (zeroed each call)
  float* st   = agg + ND_ELEMS;     // 1024 floats of stats
  float* gSumE = st;        float* gSqE = st + 128;
  float* gSumN = st + 256;  float* gSqN = st + 384;
  float* scE   = st + 512;  float* shE  = st + 640;
  float* scN   = st + 768;  float* shN  = st + 896;
  float* edge_in = st + 1024;       // [E*D]

  float* xout = (float*)d_out;            // [N*D]
  float* eout = xout + ND_ELEMS;          // [E*D]

  // 1. zero agg + stats accumulators
  {
    int n = ND_ELEMS + 1024;
    k_zero<<<(n + 255) / 256, 256, 0, stream>>>(agg, n);
  }
  // 2. node transforms (WMMA)
  {
    int blocks = (NNODE + 127) / 128;   // 391
    k_gemm_xwt<<<blocks, 256, 0, stream>>>(x, Wu, bu, Ux, NNODE);
    k_gemm_xwt<<<blocks, 256, 0, stream>>>(x, Wv, bv, Vx, NNODE);
    k_gemm_xwt<<<blocks, 256, 0, stream>>>(x, WB, bB, Bx, NNODE);
    k_gemm_xwt<<<blocks, 256, 0, stream>>>(x, WC, bC, Cx, NNODE);
  }
  // 3. edge GEMM + gather-add + BN stats (WMMA)
  k_edge_gemm<<<NEDGE / 128, 256, 0, stream>>>(ea, WA, bA, eidx, Bx, Cx,
                                               edge_in, gSumE, gSqE);
  // 4. edge BN finalize
  k_bn_finalize<<<1, 128, 0, stream>>>(gSumE, gSqE, ge_gamma, ge_beta,
                                       1.0f / (float)NEDGE, scE, shE);
  // 5. edge apply + gated message scatter
  k_edge_apply<<<NEDGE / 8, 256, 0, stream>>>(edge_in, ea, eidx, scE, shE, Vx,
                                              eout, agg);
  // 6. node BN stats
  k_node_stats<<<NNODE / 8, 256, 0, stream>>>(Ux, agg, gSumN, gSqN);
  // 7. node BN finalize
  k_bn_finalize<<<1, 128, 0, stream>>>(gSumN, gSqN, gn_gamma, gn_beta,
                                       1.0f / (float)NNODE, scN, shN);
  // 8. node apply
  k_node_apply<<<NNODE / 8, 256, 0, stream>>>(x, Ux, agg, scN, shN, xout);
}
